// MultiHeadSelfAttention_24197845746276
// MI455X (gfx1250) — matrix operand order repaired and checked by reading, not compiled, not run on hardware
//
#include <hip/hip_runtime.h>

#define NTOK   4096
#define DMODEL 1024
#define NHEADS 16
#define DKH    64

typedef __attribute__((ext_vector_type(16))) __bf16 v16bf;
typedef __attribute__((ext_vector_type(8)))  float  v8f;
typedef __attribute__((ext_vector_type(4)))  int    i32x4;

struct alignas(16) U128 { unsigned u[4]; };
struct alignas(16) F128 { float f[4]; };

union Frag {
  v16bf v;
  U128  h[2];
  unsigned u[8];
  unsigned short s[16];
};

__device__ __forceinline__ unsigned short f2bf(float f) {
  unsigned u = __builtin_bit_cast(unsigned, f);
  unsigned r = u + 0x7FFFu + ((u >> 16) & 1u);   // round-to-nearest-even
  return (unsigned short)(r >> 16);
}

__device__ __forceinline__ v8f bfwmma(v16bf a, v16bf b, v8f c) {
  // D = A(16x32 bf16) * B(32x16 bf16) + C(16x16 f32)
  return __builtin_amdgcn_wmma_f32_16x16x32_bf16(false, a, false, b, (short)0, c, false, false);
}

// --------------------------- async copy to LDS ------------------------------
#if __has_builtin(__builtin_amdgcn_global_load_async_to_lds_b128)
#define USE_ASYNC_LDS 1
#endif

__device__ __forceinline__ void copy16_to_lds(unsigned short* lds, const unsigned short* g) {
#ifdef USE_ASYNC_LDS
  __builtin_amdgcn_global_load_async_to_lds_b128(
      (__attribute__((address_space(1))) i32x4*)g,
      (__attribute__((address_space(3))) i32x4*)lds, 0, 0);
#else
  *(U128*)lds = *(const U128*)g;      // sync fallback: global load + ds_store
#endif
}

template <int N> __device__ __forceinline__ void wait_async() {
#ifdef USE_ASYNC_LDS
#if __has_builtin(__builtin_amdgcn_s_wait_asynccnt)
  __builtin_amdgcn_s_wait_asynccnt(N);
#else
  asm volatile("s_wait_asynccnt %0" :: "n"(N) : "memory");
#endif
#endif
}

// ---------------------------------------------------------------------------
// fp32 -> bf16 conversion (vectorized x4)
// ---------------------------------------------------------------------------
__global__ __launch_bounds__(256) void cvt_bf16_kernel(const float* __restrict__ src,
                                                       unsigned short* __restrict__ dst, int n4) {
  int i = blockIdx.x * 256 + threadIdx.x;
  if (i < n4) {
    F128 v = *(const F128*)(src + (size_t)i * 4);
    unsigned lo = (unsigned)f2bf(v.f[0]) | ((unsigned)f2bf(v.f[1]) << 16);
    unsigned hi = (unsigned)f2bf(v.f[2]) | ((unsigned)f2bf(v.f[3]) << 16);
    unsigned* p = (unsigned*)(dst + (size_t)i * 4);
    p[0] = lo; p[1] = hi;
  }
}

// ---------------------------------------------------------------------------
// y = A[M x K] @ W[N x K]^T + bias  -- WMMA bf16, f32 accumulate.
// Wave tile 16x64; block 8 waves = 128x64. Ping-pong fragment sets with a
// 2-step unroll: set X computes while set Y loads, then roles swap -- no
// register-copy rotation in the hot loop. Requires K % 64 == 0 (K = 1024).
// layout 0: bf16 [h][row][d] ; layout 1: bf16 [col][row] ; layout 2: f32 [row][col]
// ---------------------------------------------------------------------------
__global__ __launch_bounds__(256) void gemm_wt_kernel(const unsigned short* __restrict__ A,
                                                      const unsigned short* __restrict__ W,
                                                      const float* __restrict__ bias,
                                                      void* __restrict__ dst,
                                                      int M, int K, int NOUT, int layout) {
  const int lane = threadIdx.x & 31;
  const int wave = threadIdx.x >> 5;
  const int mw = blockIdx.x * 128 + wave * 16;
  const int n0 = blockIdx.y * 64;
  const int r15 = lane & 15;
  const int aoff = (lane < 16) ? 0 : 16;
  const int boff = (lane < 16) ? 0 : 16;

  const unsigned short* pa  = A + (size_t)(mw + r15) * K + aoff;
  const unsigned short* pb0 = W + (size_t)(n0 + 0  + r15) * K + boff;
  const unsigned short* pb1 = W + (size_t)(n0 + 16 + r15) * K + boff;
  const unsigned short* pb2 = W + (size_t)(n0 + 32 + r15) * K + boff;
  const unsigned short* pb3 = W + (size_t)(n0 + 48 + r15) * K + boff;

  v8f acc[4] = {};
  Frag xa, xb0, xb1, xb2, xb3;     // ping
  Frag ya, yb0, yb1, yb2, yb3;     // pong

#define LD_FRAGS(k0, A_, B0_, B1_, B2_, B3_)                                      \
  do {                                                                            \
    A_.h[0]  = *(const U128*)(pa  + (k0));      A_.h[1]  = *(const U128*)(pa  + (k0) + 8);  \
    B0_.h[0] = *(const U128*)(pb0 + (k0));      B0_.h[1] = *(const U128*)(pb0 + (k0) + 8);  \
    B1_.h[0] = *(const U128*)(pb1 + (k0));      B1_.h[1] = *(const U128*)(pb1 + (k0) + 8);  \
    B2_.h[0] = *(const U128*)(pb2 + (k0));      B2_.h[1] = *(const U128*)(pb2 + (k0) + 8);  \
    B3_.h[0] = *(const U128*)(pb3 + (k0));      B3_.h[1] = *(const U128*)(pb3 + (k0) + 8);  \
  } while (0)

  LD_FRAGS(0, xa, xb0, xb1, xb2, xb3);
  for (int k0 = 0; k0 < K; k0 += 64) {
    // load pong set for k0+32 while computing ping set at k0
    LD_FRAGS(k0 + 32, ya, yb0, yb1, yb2, yb3);
    __builtin_prefetch(pa + k0 + 128, 0, 1);
    acc[0] = bfwmma(xa.v, xb0.v, acc[0]);
    acc[1] = bfwmma(xa.v, xb1.v, acc[1]);
    acc[2] = bfwmma(xa.v, xb2.v, acc[2]);
    acc[3] = bfwmma(xa.v, xb3.v, acc[3]);
    // load ping set for k0+64 while computing pong set at k0+32
    if (k0 + 64 < K) LD_FRAGS(k0 + 64, xa, xb0, xb1, xb2, xb3);
    acc[0] = bfwmma(ya.v, yb0.v, acc[0]);
    acc[1] = bfwmma(ya.v, yb1.v, acc[1]);
    acc[2] = bfwmma(ya.v, yb2.v, acc[2]);
    acc[3] = bfwmma(ya.v, yb3.v, acc[3]);
  }
#undef LD_FRAGS

  const int rowb = mw + ((lane < 16) ? 0 : 8);
  for (int j = 0; j < 4; j++) {
    const int col = n0 + j * 16 + r15;
    const float b = bias[col];
    if (layout == 2) {
      float* o = (float*)dst;
      for (int r = 0; r < 8; r++) o[(size_t)(rowb + r) * NOUT + col] = acc[j][r] + b;
    } else if (layout == 0) {
      unsigned short* o = (unsigned short*)dst;
      const int h = col >> 6, d = col & 63;
      for (int r = 0; r < 8; r++)
        o[((size_t)h * M + rowb + r) * DKH + d] = f2bf(acc[j][r] + b);
    } else {
      unsigned short* o = (unsigned short*)dst;   // [col][M], contiguous 16B store
      U128 pk;
      for (int r = 0; r < 4; r++)
        pk.u[r] = (unsigned)f2bf(acc[j][2 * r] + b) | ((unsigned)f2bf(acc[j][2 * r + 1] + b) << 16);
      *(U128*)(o + (size_t)col * M + rowb) = pk;
    }
  }
}

// ---------------------------------------------------------------------------
// Flash attention, transposed (S^T = K Q^T, O^T = V^T P^T), with K/V tiles
// staged in LDS by async copies shared across the 8 waves of the block
// (8x global-traffic reduction), double-buffered against compute.
// ---------------------------------------------------------------------------
__global__ __launch_bounds__(256) void flash_attn_kernel(const unsigned short* __restrict__ Q,   // [16][4096][64]
                                                         const unsigned short* __restrict__ Km,  // [16][4096][64]
                                                         const unsigned short* __restrict__ Vt,  // [16][64][4096]
                                                         const float* __restrict__ rg,           // [4096][4096]
                                                         unsigned short* __restrict__ ctx) {     // [4096][1024]
  __shared__ unsigned short Kb[2][32 * 64];   // [key][d]
  __shared__ unsigned short Vb[2][64 * 32];   // [d][m]

  const int t = threadIdx.x;
  const int lane = t & 31;
  const int wave = t >> 5;
  const int head = blockIdx.y;
  const int r15 = lane & 15;
  const int qg = blockIdx.x * 128 + wave * 16 + r15;
  const int aoff = (lane < 16) ? 0 : 8;
  const int boff = (lane < 16) ? 0 : 8;

  const unsigned short* Qh = Q  + (size_t)head * NTOK * DKH;
  const unsigned short* Kh = Km + (size_t)head * NTOK * DKH;
  const unsigned short* Vh = Vt + (size_t)head * DKH * NTOK;

  // staging assignment: 2 async 16B chunks per thread per tile (64/wave)
  const int kkey = t >> 3, kchunk = t & 7;    // K tile: 32 rows x 8 chunks
  const int vd   = t >> 2, vchunk = t & 3;    // V tile: 64 rows x 4 chunks

  // Q as B fragments (k = d), loaded once.
  const unsigned short* pq = Qh + (size_t)qg * DKH;
  Frag bq0, bq1;
  bq0.h[0] = *(const U128*)(pq + boff);      bq0.h[1] = *(const U128*)(pq + boff + 16);
  bq1.h[0] = *(const U128*)(pq + 32 + boff); bq1.h[1] = *(const U128*)(pq + 32 + boff + 16);

  v8f o[4] = {};
  float m_run = -3.0e38f, l_run = 0.0f;
  const float scale = 0.125f;                 // 1/sqrt(64)
  const float* rgrow = rg + (size_t)qg * NTOK + aoff;

  // prologue: stage tile 0 into buffer 0
  copy16_to_lds(&Kb[0][kkey * 64 + kchunk * 8], Kh + (size_t)kkey * DKH + kchunk * 8);
  copy16_to_lds(&Vb[0][vd * 32 + vchunk * 8],   Vh + (size_t)vd * NTOK + vchunk * 8);

  const int T = NTOK / 32;
  for (int i = 0; i < T; i++) {
    const int m0 = i * 32;
    const int p = i & 1;
    if (i + 1 < T) {
      const int m1 = m0 + 32;
      copy16_to_lds(&Kb[p ^ 1][kkey * 64 + kchunk * 8], Kh + (size_t)(m1 + kkey) * DKH + kchunk * 8);
      copy16_to_lds(&Vb[p ^ 1][vd * 32 + vchunk * 8],   Vh + (size_t)vd * NTOK + m1 + vchunk * 8);
      wait_async<63>();   // in-order: >=65 oldest done => tile i (64/wave) landed
      __builtin_prefetch(rgrow + m0 + 32, 0, 1);
    } else {
      wait_async<0>();
    }
    __syncthreads();      // tile i visible to all waves

    const unsigned short* kb = Kb[p];
    const unsigned short* vb = Vb[p];

    // ---- scores^T tile: 32 keys x 16 queries ----
    v8f s0 = {}, s1 = {};
    {
      const unsigned short* pA = kb + r15 * 64 + aoff;            // keys 0-15
      Frag a; a.h[0] = *(const U128*)(pA);      a.h[1] = *(const U128*)(pA + 16);
      s0 = bfwmma(a.v, bq0.v, s0);
      a.h[0] = *(const U128*)(pA + 32); a.h[1] = *(const U128*)(pA + 48);
      s0 = bfwmma(a.v, bq1.v, s0);
      const unsigned short* pB = kb + (16 + r15) * 64 + aoff;     // keys 16-31
      a.h[0] = *(const U128*)(pB);      a.h[1] = *(const U128*)(pB + 16);
      s1 = bfwmma(a.v, bq0.v, s1);
      a.h[0] = *(const U128*)(pB + 32); a.h[1] = *(const U128*)(pB + 48);
      s1 = bfwmma(a.v, bq1.v, s1);
    }

    // ---- multiplicative RWR mask + online softmax ----
    float r0[8], r1[8];
    *(F128*)&r0[0] = *(const F128*)(rgrow + m0);
    *(F128*)&r0[4] = *(const F128*)(rgrow + m0 + 4);
    *(F128*)&r1[0] = *(const F128*)(rgrow + m0 + 16);
    *(F128*)&r1[4] = *(const F128*)(rgrow + m0 + 20);
    float p0[8], p1[8], mx = -3.0e38f;
    for (int r = 0; r < 8; r++) {
      p0[r] = s0[r] * scale * r0[r];
      p1[r] = s1[r] * scale * r1[r];
      mx = fmaxf(mx, fmaxf(p0[r], p1[r]));
    }
    mx = fmaxf(mx, __shfl_xor(mx, 16, 32));
    const float mnew = fmaxf(m_run, mx);
    const float corr = __expf(m_run - mnew);
    float ssum = 0.0f;
    for (int r = 0; r < 8; r++) {
      p0[r] = __expf(p0[r] - mnew);
      p1[r] = __expf(p1[r] - mnew);
      ssum += p0[r] + p1[r];
    }
    ssum += __shfl_xor(ssum, 16, 32);
    l_run = l_run * corr + ssum;
    m_run = mnew;

    // ---- P^T C->B layout fix: pack bf16 pairs + half-wave swap ----
    unsigned u0[4], u1[4];
    for (int j = 0; j < 4; j++) {
      u0[j] = (unsigned)f2bf(p0[2 * j]) | ((unsigned)f2bf(p0[2 * j + 1]) << 16);
      u1[j] = (unsigned)f2bf(p1[2 * j]) | ((unsigned)f2bf(p1[2 * j + 1]) << 16);
    }
    Frag bp;
    for (int j = 0; j < 4; j++) {
      unsigned x0 = (unsigned)__shfl_xor((int)u0[j], 16, 32);
      unsigned x1 = (unsigned)__shfl_xor((int)u1[j], 16, 32);
      bp.u[j]     = (lane < 16) ? u0[j] : x1;
      bp.u[4 + j] = (lane < 16) ? x0 : u1[j];
    }

    // ---- rescale accumulators, O^T += V^T @ P^T ----
    for (int j = 0; j < 4; j++)
      for (int e = 0; e < 8; e++) o[j][e] *= corr;
    for (int j = 0; j < 4; j++) {
      const unsigned short* pv = vb + (j * 16 + r15) * 32 + 2 * aoff;
      Frag av; av.h[0] = *(const U128*)pv; av.h[1] = *(const U128*)(pv + 8);
      o[j] = bfwmma(av.v, bp.v, o[j]);
    }
    __syncthreads();      // all reads of buf p done before it is re-staged
  }

  // ---- normalize, store context row fragment (bf16, contiguous along d) ----
  const float inv = 1.0f / l_run;
  const int rowd = (lane < 16) ? 0 : 8;
  unsigned short* pc = ctx + (size_t)qg * DMODEL + head * DKH;
  for (int j = 0; j < 4; j++) {
    U128 pk;
    for (int r = 0; r < 4; r++)
      pk.u[r] = (unsigned)f2bf(o[j][2 * r] * inv) | ((unsigned)f2bf(o[j][2 * r + 1] * inv) << 16);
    *(U128*)(pc + j * 16 + rowd) = pk;
  }
}

// ---------------------------------------------------------------------------
extern "C" void kernel_launch(void* const* d_in, const int* in_sizes, int n_in,
                              void* d_out, int out_size, void* d_ws, size_t ws_size,
                              hipStream_t stream) {
  (void)in_sizes; (void)n_in; (void)out_size; (void)ws_size;
  const float* H  = (const float*)d_in[0];
  const float* rg = (const float*)d_in[1];
  const float* Wq = (const float*)d_in[2];  const float* bq = (const float*)d_in[3];
  const float* Wk = (const float*)d_in[4];  const float* bk = (const float*)d_in[5];
  const float* Wv = (const float*)d_in[6];  const float* bv = (const float*)d_in[7];
  const float* Wo = (const float*)d_in[8];  const float* bo = (const float*)d_in[9];
  float* out = (float*)d_out;

  unsigned short* Hbf  = (unsigned short*)d_ws;
  unsigned short* Wqbf = Hbf  + (size_t)NTOK * DMODEL;
  unsigned short* Wkbf = Wqbf + (size_t)DMODEL * DMODEL;
  unsigned short* Wvbf = Wkbf + (size_t)DMODEL * DMODEL;
  unsigned short* Wobf = Wvbf + (size_t)DMODEL * DMODEL;
  unsigned short* Qbf  = Wobf + (size_t)DMODEL * DMODEL;   // [h][n][64]
  unsigned short* Kbf  = Qbf  + (size_t)NTOK * DMODEL;     // [h][n][64]
  unsigned short* Vtbf = Kbf  + (size_t)NTOK * DMODEL;     // [h][64][n]
  unsigned short* Cbf  = Vtbf + (size_t)NTOK * DMODEL;     // [n][1024]

  const int nH4 = (NTOK * DMODEL) / 4, nW4 = (DMODEL * DMODEL) / 4;
  cvt_bf16_kernel<<<(nH4 + 255) / 256, 256, 0, stream>>>(H,  Hbf,  nH4);
  cvt_bf16_kernel<<<(nW4 + 255) / 256, 256, 0, stream>>>(Wq, Wqbf, nW4);
  cvt_bf16_kernel<<<(nW4 + 255) / 256, 256, 0, stream>>>(Wk, Wkbf, nW4);
  cvt_bf16_kernel<<<(nW4 + 255) / 256, 256, 0, stream>>>(Wv, Wvbf, nW4);
  cvt_bf16_kernel<<<(nW4 + 255) / 256, 256, 0, stream>>>(Wo, Wobf, nW4);

  dim3 gg(NTOK / 128, DMODEL / 64);
  gemm_wt_kernel<<<gg, 256, 0, stream>>>(Hbf, Wqbf, bq, Qbf,  NTOK, DMODEL, DMODEL, 0);
  gemm_wt_kernel<<<gg, 256, 0, stream>>>(Hbf, Wkbf, bk, Kbf,  NTOK, DMODEL, DMODEL, 0);
  gemm_wt_kernel<<<gg, 256, 0, stream>>>(Hbf, Wvbf, bv, Vtbf, NTOK, DMODEL, DMODEL, 1);

  flash_attn_kernel<<<dim3(NTOK / 128, NHEADS), 256, 0, stream>>>(Qbf, Kbf, Vtbf, rg, Cbf);

  gemm_wt_kernel<<<gg, 256, 0, stream>>>(Cbf, Wobf, bo, out, NTOK, DMODEL, DMODEL, 2);
}
